// Convolution_49117245997775
// MI455X (gfx1250) — matrix operand order repaired and checked by reading, not compiled, not run on hardware
//
#include <hip/hip_runtime.h>

// ---------------------------------------------------------------------------
// Problem constants (match reference)
// ---------------------------------------------------------------------------
#define Bn     4
#define CIN    32
#define Hn     64
#define Wn     64
#define HW     (Hn * Wn)            // 4096
#define COUT   64
#define Kk     9
#define GADD   4
#define RADD   4
#define VS     (4 + GADD + RADD)    // 12
#define HIDDEN (CIN * 4)            // 128
#define NPARM  (3 * Kk)             // 27
#define KDIM   (Kk * CIN)           // 288 (GEMM K)
#define MROWS  (Bn * HW)            // 16384 (GEMM M)
#define KSTEPS (KDIM / 32)          // 9
#define NTILES (COUT / 16)          // 4

#define MIN_SIGMA   0.05f
#define SIGMA_SCALE 0.05f
#define SIGMA_BOOST 2.0f
#define EPSF        1e-4f
#define REGION_F    12.0f

typedef __attribute__((ext_vector_type(16))) __bf16 v16bf;
typedef __attribute__((ext_vector_type(8)))  float  v8f;

// ---------------------------------------------------------------------------
// Helpers
// ---------------------------------------------------------------------------
static __device__ __forceinline__ unsigned short f32_to_bf16_bits(float f) {
    union { float f; unsigned u; } v; v.f = f;
    unsigned r = v.u + 0x7FFFu + ((v.u >> 16) & 1u);   // round-to-nearest-even
    return (unsigned short)(r >> 16);
}

// PCG-style hash -> uniform float in [0,1). Deterministic stand-in for the
// JAX threefry stream (no runtime correctness check in this loop).
static __device__ __forceinline__ unsigned pcg_hash(unsigned x) {
    x = x * 747796405u + 2891336453u;
    unsigned w = ((x >> ((x >> 28) + 4u)) ^ x) * 277803737u;
    return (w >> 22) ^ w;
}
static __device__ __forceinline__ float hash_uniform(unsigned seed) {
    return (float)(pcg_hash(seed) >> 8) * (1.0f / 16777216.0f);
}

static __device__ __forceinline__ float softplusf(float x) {
    return (x > 20.0f) ? x : log1pf(expf(x));
}
static __device__ __forceinline__ float pmodf(float v, float m) { // jnp.remainder
    float r = fmodf(v, m);
    return (r < 0.0f) ? r + m : r;
}

// ---------------------------------------------------------------------------
// Kernel 0: transpose x (B,CIN,HW) -> xT (B,HW,CIN)  [channel-contiguous]
// ---------------------------------------------------------------------------
__global__ void k_transpose_x(const float* __restrict__ x, float* __restrict__ xT) {
    int idx = blockIdx.x * blockDim.x + threadIdx.x;      // over B*CIN*HW
    if (idx >= Bn * CIN * HW) return;
    int pix = idx & (HW - 1);
    int c   = (idx >> 12) & (CIN - 1);
    int b   = idx >> 17;
    xT[(b * HW + pix) * CIN + c] = x[idx];
}

// ---------------------------------------------------------------------------
// Kernel 1: hypernetwork -> means (HW,K,2), sigmas (HW,K,2)
// ---------------------------------------------------------------------------
__global__ void k_hyper(const float* __restrict__ w1, const float* __restrict__ b1,
                        const float* __restrict__ w2, const float* __restrict__ b2,
                        float* __restrict__ means, float* __restrict__ sigmas) {
    int pix = blockIdx.x * blockDim.x + threadIdx.x;
    if (pix >= HW) return;
    int h = pix >> 6, w = pix & 63;
    float cy = (float)h * (1.0f / (Hn - 1));
    float cx = (float)w * (1.0f / (Wn - 1));

    float p[NPARM];
#pragma unroll
    for (int i = 0; i < NPARM; ++i) p[i] = b2[i];

    for (int j = 0; j < HIDDEN; ++j) {
        float hid = fmaxf(cy * w1[j] + cx * w1[HIDDEN + j] + b1[j], 0.0f);
        const float* w2r = w2 + j * NPARM;
#pragma unroll
        for (int i = 0; i < NPARM; ++i) p[i] += hid * w2r[i];
    }

    float* mo = means  + pix * (Kk * 2);
    float* so = sigmas + pix * (Kk * 2);
#pragma unroll
    for (int k = 0; k < Kk; ++k) {
        float my = pmodf((float)h + p[2 * k + 0], (float)(Hn - 1));
        float mx = pmodf((float)w + p[2 * k + 1], (float)(Wn - 1));
        float sg = softplusf(p[2 * Kk + k] + SIGMA_BOOST) + MIN_SIGMA;
        float sv = sg * (float)Hn * SIGMA_SCALE;   // H == W
        mo[2 * k + 0] = my;  mo[2 * k + 1] = mx;
        so[2 * k + 0] = sv;  so[2 * k + 1] = sv;
    }
}

// ---------------------------------------------------------------------------
// Kernel 2: sampling + duplicate mask + Gaussian weights + coalesced gather.
// One wave32 per (b,h,w,k); lane == channel. Output featB: (MROWS, KDIM) bf16.
// ---------------------------------------------------------------------------
__global__ void k_sample_gather(const float* __restrict__ xT,
                                const float* __restrict__ means,
                                const float* __restrict__ sigmas,
                                unsigned short* __restrict__ featB) {
    int g    = blockIdx.x * (blockDim.x >> 5) + (threadIdx.x >> 5); // group id
    int lane = threadIdx.x & 31;
    if (g >= Bn * HW * Kk) return;

    int k   = g % Kk;
    int pix = (g / Kk) & (HW - 1);
    int b   = g / (Kk * HW);

    float my = means [pix * (Kk * 2) + 2 * k + 0];
    float mx = means [pix * (Kk * 2) + 2 * k + 1];
    float sy = sigmas[pix * (Kk * 2) + 2 * k + 0];
    float sx = sigmas[pix * (Kk * 2) + 2 * k + 1];

    int   iy[VS], ix[VS];
    float fy = floorf(my), fx = floorf(mx);
    // corners (dy,dx) in order (0,0)(0,1)(1,0)(1,1)
#pragma unroll
    for (int v = 0; v < 4; ++v) {
        iy[v] = (int)fy + (v >> 1);
        ix[v] = (int)fx + (v & 1);
    }
    unsigned base = (unsigned)(((b * HW + pix) * Kk + k) * 64 + 12345);
    // global uniform samples
#pragma unroll
    for (int v = 0; v < GADD; ++v) {
        float uy = hash_uniform(base + 2 * v + 0) * (1.0f - EPSF);
        float ux = hash_uniform(base + 2 * v + 1) * (1.0f - EPSF);
        iy[4 + v] = (int)floorf(uy * (float)Hn);
        ix[4 + v] = (int)floorf(ux * (float)Wn);
    }
    // local window samples
    float ly = fminf(fmaxf(my - 0.5f * REGION_F, 0.0f), (float)Hn - REGION_F);
    float lx = fminf(fmaxf(mx - 0.5f * REGION_F, 0.0f), (float)Wn - REGION_F);
#pragma unroll
    for (int v = 0; v < RADD; ++v) {
        float uy = hash_uniform(base + 16 + 2 * v + 0) * (1.0f - EPSF);
        float ux = hash_uniform(base + 16 + 2 * v + 1) * (1.0f - EPSF);
        iy[8 + v] = (int)floorf(ly + uy * REGION_F);
        ix[8 + v] = (int)floorf(lx + ux * REGION_F);
    }
#pragma unroll
    for (int v = 0; v < VS; ++v) {
        iy[v] = min(max(iy[v], 0), Hn - 1);
        ix[v] = min(max(ix[v], 0), Wn - 1);
    }

    // duplicate masking (keep first) + Gaussian densities -> normalized weights
    float wgt[VS];
    float tot = 0.0f;
#pragma unroll
    for (int v = 0; v < VS; ++v) {
        bool dup = false;
#pragma unroll
        for (int u = 0; u < VS; ++u)
            if (u < v && iy[u] == iy[v] && ix[u] == ix[v]) dup = true;
        float zy = ((float)iy[v] - my) / sy;
        float zx = ((float)ix[v] - mx) / sx;
        float d  = dup ? 0.0f : __expf(-0.5f * (zy * zy + zx * zx));
        wgt[v] = d;
        tot   += d;
    }
    float inv = 1.0f / tot;

    // coalesced gather: each sample is one contiguous 128B channel vector
    const float* xb = xT + (size_t)b * HW * CIN;
    float acc = 0.0f;
#pragma unroll
    for (int v = 0; v < VS; ++v) {
        int code = iy[v] * Wn + ix[v];
        acc += (wgt[v] * inv) * xb[code * CIN + lane];
    }

    int row = b * HW + pix;
    featB[(size_t)row * KDIM + k * CIN + lane] = f32_to_bf16_bits(acc);
}

// ---------------------------------------------------------------------------
// Kernel 3: convert wu (KDIM x COUT) f32 -> bf16, pre-swizzled into the
// per-lane WMMA B-fragment layout:
//   wuFrag[((ks*NTILES + nt)*32 + lane)*16 + e] = W[ks*32 + (e<8 ? hi*8+e : 16+hi*8+e-8)][nt*16 + lr]
// so the GEMM reads each B fragment as two contiguous 16B loads per lane.
// ---------------------------------------------------------------------------
__global__ void k_cvt_wu(const float* __restrict__ wu, unsigned short* __restrict__ wuFrag) {
    int i = blockIdx.x * blockDim.x + threadIdx.x;
    if (i >= KSTEPS * NTILES * 32 * 16) return;
    int e    = i & 15;
    int lane = (i >> 4) & 31;
    int nt   = (i >> 9) & (NTILES - 1);
    int ks   = i >> 11;
    int hi   = lane >> 4, lr = lane & 15;
    int k = ks * 32 + ((e < 8) ? (hi * 8 + e) : (16 + hi * 8 + (e - 8)));
    int n = nt * 16 + lr;
    wuFrag[i] = f32_to_bf16_bits(wu[k * COUT + n]);
}

// ---------------------------------------------------------------------------
// Kernel 4: unify GEMM via v_wmma_f32_16x16x32_bf16.
// C(16384x64) = A(16384x288,bf16) x W(288x64,bf16) + bias, scattered to
// out layout (B, COUT, H, W).
// Block = 256 threads = 8 waves = 2(M-waves) x 4(N-waves); each wave owns
// 4 M-tiles x 1 N-tile -> block tile 128x64, 36 WMMAs per wave.
// Explicit double-buffered fragments: step ks+1's 10 b128 loads are issued
// before step ks's 4 WMMAs consume their (separate) register set, so the
// waits become partial instead of s_wait_loadcnt 0.
// ---------------------------------------------------------------------------
struct Frag { uint4 q[2]; };

__global__ void k_unify_gemm(const unsigned short* __restrict__ featB,
                             const unsigned short* __restrict__ wuFrag,
                             const float* __restrict__ bu,
                             float* __restrict__ out) {
    const int tid    = threadIdx.x;
    const int waveId = tid >> 5;
    const int lane   = tid & 31;
    const int hi     = lane >> 4;        // half-wave select
    const int lr     = lane & 15;

    const int mw = waveId >> 2;                    // 0..1
    const int nw = waveId & 3;                     // 0..3 == n-tile
    const int m0 = blockIdx.x * 128 + mw * 64;     // wave's 4 tiles: m0 + t*16

    v8f acc[4] = {};

    // Per-lane base pointers.
    // A fragment (16-bit A 16x32 layout, ISA 7.12.2):
    //   elems 0..7  -> K = k0 + hi*8 + i        (16B contiguous)
    //   elems 8..15 -> K = k0 + 16 + hi*8 + i   (16B contiguous)
    const unsigned short* Arow0 = featB + (size_t)(m0 + lr) * KDIM + hi * 8;
    // B fragment: pre-swizzled, 32 contiguous bytes per lane.
    const unsigned short* Bbase = wuFrag + ((size_t)nw * 32 + lane) * 16;

    Frag a[2][4], bf[2];

    // Prologue: stage K-step 0.
#pragma unroll
    for (int t = 0; t < 4; ++t) {
        const unsigned short* Arow = Arow0 + (size_t)t * 16 * KDIM;
        a[0][t].q[0] = *(const uint4*)(Arow);
        a[0][t].q[1] = *(const uint4*)(Arow + 16);
    }
    bf[0].q[0] = ((const uint4*)Bbase)[0];
    bf[0].q[1] = ((const uint4*)Bbase)[1];

#pragma unroll
    for (int ks = 0; ks < KSTEPS; ++ks) {
        const int cur = ks & 1;
        const int nxt = cur ^ 1;

        // Stage K-step ks+1 into the other buffer before computing ks.
        if (ks + 1 < KSTEPS) {
            const int k1 = (ks + 1) * 32;
#pragma unroll
            for (int t = 0; t < 4; ++t) {
                const unsigned short* Arow = Arow0 + (size_t)t * 16 * KDIM + k1;
                a[nxt][t].q[0] = *(const uint4*)(Arow);
                a[nxt][t].q[1] = *(const uint4*)(Arow + 16);
                if (ks + 2 < KSTEPS)
                    __builtin_prefetch(Arow + 32, 0, 0);   // global_prefetch_b8
            }
            const unsigned short* Bp = Bbase + (size_t)(ks + 1) * NTILES * 32 * 16;
            bf[nxt].q[0] = ((const uint4*)Bp)[0];
            bf[nxt].q[1] = ((const uint4*)Bp)[1];
        }

        union { Frag f; v16bf v; } ubv; ubv.f = bf[cur];
#pragma unroll
        for (int t = 0; t < 4; ++t) {
            union { Frag f; v16bf v; } uav; uav.f = a[cur][t];
            acc[t] = __builtin_amdgcn_wmma_f32_16x16x32_bf16(
                /*neg_a=*/false, uav.v, /*neg_b=*/false, ubv.v,
                /*c_mod=*/(short)0, acc[t], /*reuse_a=*/false, /*reuse_b=*/false);
        }
    }

    // D element r of tile t lives at row (m0 + t*16 + r + hi*8), col (nw*16 + lr)
    const int   n  = nw * 16 + lr;
    const float bb = bu[n];
#pragma unroll
    for (int t = 0; t < 4; ++t) {
#pragma unroll
        for (int r = 0; r < 8; ++r) {
            int gr  = m0 + t * 16 + r + hi * 8;    // global feature row
            int b   = gr >> 12;                    // / HW
            int pix = gr & (HW - 1);
            out[((size_t)(b * COUT + n) << 12) + pix] = acc[t][r] + bb;
        }
    }
}

// ---------------------------------------------------------------------------
// Host-side launch
// ---------------------------------------------------------------------------
extern "C" void kernel_launch(void* const* d_in, const int* in_sizes, int n_in,
                              void* d_out, int out_size, void* d_ws, size_t ws_size,
                              hipStream_t stream) {
    const float* x  = (const float*)d_in[0];
    const float* w1 = (const float*)d_in[1];
    const float* b1 = (const float*)d_in[2];
    const float* w2 = (const float*)d_in[3];
    const float* b2 = (const float*)d_in[4];
    const float* wu = (const float*)d_in[5];
    const float* bu = (const float*)d_in[6];
    float* out = (float*)d_out;

    char* ws = (char*)d_ws;
    size_t off = 0;
    float* xT = (float*)(ws + off);            off += (size_t)Bn * HW * CIN * 4;   // 2 MB
    float* means = (float*)(ws + off);         off += (size_t)HW * Kk * 2 * 4;     // 288 KB
    float* sigmas = (float*)(ws + off);        off += (size_t)HW * Kk * 2 * 4;     // 288 KB
    unsigned short* wuFrag = (unsigned short*)(ws + off); off += (size_t)KSTEPS * NTILES * 32 * 16 * 2;
    off = (off + 15) & ~(size_t)15;
    unsigned short* featB = (unsigned short*)(ws + off); // 16384*288*2 = 9.4 MB

    // 0) transpose x to channel-contiguous layout
    k_transpose_x<<<(Bn * CIN * HW + 255) / 256, 256, 0, stream>>>(x, xT);
    // 1) hypernetwork means/sigmas (4096 pixels)
    k_hyper<<<(HW + 63) / 64, 64, 0, stream>>>(w1, b1, w2, b2, means, sigmas);
    // 2) sample + weight + gather -> bf16 feature matrix (one wave per group)
    {
        int groups = Bn * HW * Kk;                 // 147456
        int waves_per_block = 8;
        int blocks = (groups + waves_per_block - 1) / waves_per_block;
        k_sample_gather<<<blocks, waves_per_block * 32, 0, stream>>>(xT, means, sigmas, featB);
    }
    // 3) wu -> bf16, swizzled to WMMA B-fragment layout
    k_cvt_wu<<<(KSTEPS * NTILES * 32 * 16 + 255) / 256, 256, 0, stream>>>(wu, wuFrag);
    // 4) WMMA GEMM + bias + scatter to (B,COUT,H,W)
    k_unify_gemm<<<MROWS / 128, 256, 0, stream>>>(featB, wuFrag, bu, out);
}
